// LSTMTemporalClassifier_87789131530975
// MI455X (gfx1250) — matrix operand order repaired and checked by reading, not compiled, not run on hardware
//
#include <hip/hip_runtime.h>

// ---------------------------------------------------------------------------
// LSTM temporal classifier for MI455X (gfx1250), wave32 + WMMA f16 path.
//
//   prep:   fold embedding into layer-0 input weights (W_comb = W_ih0@W_emb),
//           pack all recurrent weights into f16 WMMA B-fragment layout in ws.
//   pass0:  layer-0 LSTM. W_comb + W_hh0 resident in LDS (f16 fragments).
//           128 workgroups x 16 batch rows, 8 waves each (wave owns 64 gate
//           cols). Per step: v_wmma_f32_16x16x32_f16 GEMM (zero-C init, bias
//           folded into activations) -> LDS gates -> per-thread activations
//           (c in registers across T) -> h to LDS + NT-streamed h tile to HBM.
//           x values for step t+1 prefetched into registers during step t.
//   pass1:  layer-1 LSTM (h0 tiles NT-prefetched one step ahead), W_ih1 +
//           W_hh1 in LDS (256KB), then fused FC head 128->64->1 -> logits.
// ---------------------------------------------------------------------------

typedef __attribute__((ext_vector_type(16))) _Float16 v16h;
typedef __attribute__((ext_vector_type(8)))  _Float16 v8h;
typedef __attribute__((ext_vector_type(8)))  float    v8f;
typedef __attribute__((ext_vector_type(4)))  float    v4f;

namespace {
constexpr int kB = 2048, kT = 576;
constexpr int kNWG = kB / 16;            // 128 workgroups, 16 batch rows each
constexpr int kNT  = 512 / 16;           // 32 n-tiles over the 512 gate cols
constexpr int kKC  = 128 / 32;           // 4 K-chunks of 32 over hidden dim

constexpr int WC_HALF = kNT * 32 * 16;        // packed W_comb (K padded 15->32)
constexpr int WM_HALF = kNT * kKC * 32 * 16;  // packed 512x128 matrix

constexpr size_t OFF_WC   = 0;
constexpr size_t OFF_WH0  = OFF_WC  + (size_t)WC_HALF * 2;
constexpr size_t OFF_WI1  = OFF_WH0 + (size_t)WM_HALF * 2;
constexpr size_t OFF_WH1  = OFF_WI1 + (size_t)WM_HALF * 2;
constexpr size_t OFF_B0   = OFF_WH1 + (size_t)WM_HALF * 2;
constexpr size_t OFF_B1   = OFF_B0  + 512 * 4;
constexpr size_t OFF_OUT0 = OFF_B1  + 512 * 4;   // f16 [T][128 tiles][16][128]

constexpr int HSTR = 136;  // LDS h row stride (halves): 16B-aligned rows, no bank conflict
constexpr int XSTR = 40;   // LDS x row stride (halves)
constexpr int GSTR = 20;   // LDS gates [N][GSTR] floats: 16B-aligned v4f stores
}

__device__ __forceinline__ float dev_sigm(float x) {
  return 1.0f / (1.0f + __expf(-x));
}
__device__ __forceinline__ float dev_tanh(float x) {
  return 2.0f / (1.0f + __expf(-2.0f * x)) - 1.0f;
}

// A-fragment (16x32 f16) load from a row-major LDS row pointer.
// Layout (matches packed B): lanes 0-15 (M=lane) hold K {0..15}; lanes 16-31 hold
// K {16..31}. klo = (lane>=16)?8:0.
__device__ __forceinline__ v16h load_a_frag(const _Float16* row, int kb, int klo) {
  v8h lo = *(const v8h*)(row + kb + 2 * klo);
  v8h hi = *(const v8h*)(row + kb + 2 * klo + 8);
  v16h r;
#pragma unroll
  for (int i = 0; i < 8; ++i) { r[i] = lo[i]; r[8 + i] = hi[i]; }
  return r;
}

__device__ __forceinline__ v8f wmma_f16(v16h a, v16h b, v8f c) {
  return __builtin_amdgcn_wmma_f32_16x16x32_f16(false, a, false, b, (short)0, c,
                                                false, false);
}

__device__ __forceinline__ v8f wmma_f16_z(v16h a, v16h b) {
  v8f z = {0.f, 0.f, 0.f, 0.f, 0.f, 0.f, 0.f, 0.f};
  return __builtin_amdgcn_wmma_f32_16x16x32_f16(false, a, false, b, (short)0, z,
                                                false, false);
}

// ---------------------------------------------------------------------------
// Prep: fold embedding, pack weights into B-fragment f16 layout, build biases.
// B-fragment element (gnt, kc, lane, e): N = gnt*16 + (lane&15),
//                                        K = kc*32 + ((lane>>4)<<4) + e.
// ---------------------------------------------------------------------------
__global__ void lstm_prep(const float* __restrict__ W_emb,  // (32,15)
                          const float* __restrict__ b_emb,  // (32)
                          const float* __restrict__ W_ih0,  // (512,32)
                          const float* __restrict__ W_hh0,  // (512,128)
                          const float* __restrict__ b_ih0,
                          const float* __restrict__ b_hh0,
                          const float* __restrict__ W_ih1,  // (512,128)
                          const float* __restrict__ W_hh1,  // (512,128)
                          const float* __restrict__ b_ih1,
                          const float* __restrict__ b_hh1,
                          char* __restrict__ ws) {
  int tid = blockIdx.x * blockDim.x + threadIdx.x;
  _Float16* wc  = (_Float16*)(ws + OFF_WC);
  _Float16* wm0 = (_Float16*)(ws + OFF_WH0);
  float*    b0  = (float*)(ws + OFF_B0);
  float*    b1  = (float*)(ws + OFF_B1);

  if (tid < WC_HALF) {
    // W_comb[n][k] = sum_j W_ih0[n][j] * W_emb[j][k], K padded 15 -> 32
    int e = tid & 15, lane = (tid >> 4) & 31, gnt = tid >> 9;
    int N = gnt * 16 + (lane & 15);
    int K = ((lane >> 4) << 4) + e;
    float v = 0.0f;
    if (K < 15) {
      for (int j = 0; j < 32; ++j) v += W_ih0[N * 32 + j] * W_emb[j * 15 + K];
    }
    wc[tid] = (_Float16)v;
  } else if (tid < WC_HALF + 3 * WM_HALF) {
    int p = tid - WC_HALF;
    int mat = p / WM_HALF;
    p -= mat * WM_HALF;
    int e = p & 15, lane = (p >> 4) & 31, kc = (p >> 9) & 3, gnt = p >> 11;
    int N = gnt * 16 + (lane & 15);
    int K = kc * 32 + ((lane >> 4) << 4) + e;
    const float* W = (mat == 0) ? W_hh0 : ((mat == 1) ? W_ih1 : W_hh1);
    wm0[(size_t)mat * WM_HALF + p] = (_Float16)W[N * 128 + K];
  } else if (tid < WC_HALF + 3 * WM_HALF + 512) {
    int n = tid - (WC_HALF + 3 * WM_HALF);
    float v = b_ih0[n] + b_hh0[n];
    for (int j = 0; j < 32; ++j) v += W_ih0[n * 32 + j] * b_emb[j];
    b0[n] = v;
  } else if (tid < WC_HALF + 3 * WM_HALF + 1024) {
    int n = tid - (WC_HALF + 3 * WM_HALF + 512);
    b1[n] = b_ih1[n] + b_hh1[n];
  }
}

// ---------------------------------------------------------------------------
// Pass 0: layer-0 LSTM over all T for one 16-row batch tile.
// ---------------------------------------------------------------------------
__global__ __launch_bounds__(256) void lstm_pass0(const float* __restrict__ x,
                                                  char* __restrict__ ws) {
  extern __shared__ char smem[];
  _Float16* sWc = (_Float16*)smem;               // WC_HALF
  _Float16* sWh = sWc + WC_HALF;                 // WM_HALF
  _Float16* sX  = sWh + WM_HALF;                 // 16*XSTR
  _Float16* sH  = sX + 16 * XSTR;                // 16*HSTR
  float*    sG  = (float*)(sH + 16 * HSTR);      // 512*GSTR
  float*    sB  = sG + 512 * GSTR;               // 512 bias

  const int tid  = threadIdx.x;
  const int lane = tid & 31;
  const int wv   = tid >> 5;          // 8 waves; wave owns gate cols [wv*64, wv*64+64)
  const int wg   = blockIdx.x;
  const int bm   = wg * 16;
  const int mrow = lane & 15;
  const int khi  = lane >> 4;
  const int klo  = khi * 8;

  // Stage packed weights (W_comb + W_hh0, contiguous in ws) into LDS.
  {
    const v8h* src = (const v8h*)(ws + OFF_WC);
    v8h* dst = (v8h*)sWc;
    for (int i = tid; i < (WC_HALF + WM_HALF) / 8; i += 256) dst[i] = src[i];
  }
  for (int i = tid; i < 16 * XSTR; i += 256) sX[i] = (_Float16)0.0f;
  for (int i = tid; i < 16 * HSTR; i += 256) sH[i] = (_Float16)0.0f;
  {
    const float* b0g = (const float*)(ws + OFF_B0);
    for (int i = tid; i < 512; i += 256) sB[i] = b0g[i];
  }

  _Float16* out0 = (_Float16*)(ws + OFF_OUT0);

  float creg[8];
#pragma unroll
  for (int r = 0; r < 8; ++r) creg[r] = 0.0f;

  // Software-pipelined x prefetch: value for step t+1 loaded during step t.
  const int xm = tid / 15, xk = tid - xm * 15;   // valid when tid < 240
  float xv = 0.0f;
  if (tid < 240) {
    xv = __builtin_nontemporal_load(&x[((size_t)(bm + xm) * kT + 0) * 15 + xk]);
  }

  __syncthreads();

  for (int t = 0; t < kT; ++t) {
    // Stage current x tile; issue NT load for next step.
    if (tid < 240) {
      sX[xm * XSTR + xk] = (_Float16)xv;
      int tn = (t + 1 < kT) ? (t + 1) : t;
      xv = __builtin_nontemporal_load(&x[((size_t)(bm + xm) * kT + tn) * 15 + xk]);
    }
    __syncthreads();

    // gates = x_t @ W_comb^T + h @ W_hh0^T   (bias added in activation)
    v8f acc[4];
    {
      v16h ax = load_a_frag(sX + mrow * XSTR, 0, klo);
#pragma unroll
      for (int nt = 0; nt < 4; ++nt) {
        v16h bf = *(const v16h*)(sWc + ((size_t)(wv * 4 + nt) * 32 + lane) * 16);
        acc[nt] = wmma_f16_z(ax, bf);     // zero-C initializer
      }
#pragma unroll
      for (int kc = 0; kc < 4; ++kc) {
        v16h ah = load_a_frag(sH + mrow * HSTR, kc * 32, klo);
#pragma unroll
        for (int nt = 0; nt < 4; ++nt) {
          v16h bf = *(const v16h*)(sWh + ((size_t)((wv * 4 + nt) * 4 + kc) * 32 + lane) * 16);
          acc[nt] = wmma_f16(ah, bf, acc[nt]);
        }
      }
    }
    // Store gates transposed [N][GSTR]; C elem v -> (M = v + 8*khi, N = n0+mrow)
#pragma unroll
    for (int nt = 0; nt < 4; ++nt) {
      int N = wv * 64 + nt * 16 + mrow;
      float* gp = sG + N * GSTR + khi * 8;
      v4f lo = {acc[nt][0], acc[nt][1], acc[nt][2], acc[nt][3]};
      v4f hi = {acc[nt][4], acc[nt][5], acc[nt][6], acc[nt][7]};
      *(v4f*)(gp) = lo;
      *(v4f*)(gp + 4) = hi;
    }
    __syncthreads();

    // Elementwise LSTM cell; c in registers across all T.
    _Float16* otile = out0 + (((size_t)t * kNWG + wg) << 11);
#pragma unroll
    for (int r = 0; r < 8; ++r) {
      int e = r * 256 + tid;
      int m = e >> 7, j = e & 127;
      float gi = sG[(j)       * GSTR + m] + sB[j];
      float gf = sG[(128 + j) * GSTR + m] + sB[128 + j];
      float gg = sG[(256 + j) * GSTR + m] + sB[256 + j];
      float go = sG[(384 + j) * GSTR + m] + sB[384 + j];
      float iv = dev_sigm(gi), fv = dev_sigm(gf);
      float gv = dev_tanh(gg), ov = dev_sigm(go);
      float c = fv * creg[r] + iv * gv;
      creg[r] = c;
      float h = ov * dev_tanh(c);
      sH[m * HSTR + j] = (_Float16)h;
      __builtin_nontemporal_store((_Float16)h, &otile[e]);  // NT stream to HBM
    }
    // Next iteration's first barrier orders sH/sG reuse.
  }
}

// ---------------------------------------------------------------------------
// Pass 1: layer-1 LSTM + FC head.
// ---------------------------------------------------------------------------
__global__ __launch_bounds__(256) void lstm_pass1(char* __restrict__ ws,
                                                  const float* __restrict__ Wfc1,
                                                  const float* __restrict__ bfc1,
                                                  const float* __restrict__ Wfc2,
                                                  const float* __restrict__ bfc2,
                                                  float* __restrict__ out) {
  extern __shared__ char smem[];
  _Float16* sWi = (_Float16*)smem;               // WM_HALF (W_ih1 packed)
  _Float16* sWh = sWi + WM_HALF;                 // WM_HALF (W_hh1 packed)
  _Float16* sH0 = sWh + WM_HALF;                 // 16*HSTR (layer-0 h tile)
  _Float16* sH  = sH0 + 16 * HSTR;               // 16*HSTR (layer-1 h)
  float*    sG  = (float*)(sH + 16 * HSTR);      // 512*GSTR
  float*    sB  = sG + 512 * GSTR;               // 512 bias

  const int tid  = threadIdx.x;
  const int lane = tid & 31;
  const int wv   = tid >> 5;
  const int wg   = blockIdx.x;
  const int bm   = wg * 16;
  const int mrow = lane & 15;
  const int khi  = lane >> 4;
  const int klo  = khi * 8;

  {
    const v8h* src = (const v8h*)(ws + OFF_WI1);
    v8h* dst = (v8h*)sWi;
    for (int i = tid; i < (2 * WM_HALF) / 8; i += 256) dst[i] = src[i];
  }
  for (int i = tid; i < 16 * HSTR; i += 256) sH[i] = (_Float16)0.0f;
  {
    const float* b1g = (const float*)(ws + OFF_B1);
    for (int i = tid; i < 512; i += 256) sB[i] = b1g[i];
  }

  const _Float16* out0 = (const _Float16*)(ws + OFF_OUT0);

  float creg[8], hreg[8];
#pragma unroll
  for (int r = 0; r < 8; ++r) { creg[r] = 0.0f; hreg[r] = 0.0f; }

  // Software-pipelined h0-tile prefetch (NT): tile for step t+1 in registers.
  const int sm = tid >> 4, sj = (tid & 15) * 8;
  v8h hv = __builtin_nontemporal_load(
      (const v8h*)(out0 + (((size_t)0 * kNWG + wg) << 11)) + tid);

  __syncthreads();

  for (int t = 0; t < kT; ++t) {
    // Stage current h0 tile into padded LDS rows; prefetch next tile.
    *(v8h*)(sH0 + sm * HSTR + sj) = hv;
    {
      size_t tn = (t + 1 < kT) ? (size_t)(t + 1) : (size_t)t;
      hv = __builtin_nontemporal_load(
          (const v8h*)(out0 + ((tn * kNWG + wg) << 11)) + tid);
    }
    __syncthreads();

    // gates = h0_t @ W_ih1^T + h1 @ W_hh1^T   (bias added in activation)
    v8f acc[4];
    {
      v16h a0 = load_a_frag(sH0 + mrow * HSTR, 0, klo);
      v16h a1 = load_a_frag(sH  + mrow * HSTR, 0, klo);
#pragma unroll
      for (int nt = 0; nt < 4; ++nt) {
        v16h bi = *(const v16h*)(sWi + ((size_t)((wv * 4 + nt) * 4 + 0) * 32 + lane) * 16);
        acc[nt] = wmma_f16_z(a0, bi);     // zero-C initializer
        v16h bh = *(const v16h*)(sWh + ((size_t)((wv * 4 + nt) * 4 + 0) * 32 + lane) * 16);
        acc[nt] = wmma_f16(a1, bh, acc[nt]);
      }
    }
#pragma unroll
    for (int kc = 1; kc < 4; ++kc) {
      v16h a0 = load_a_frag(sH0 + mrow * HSTR, kc * 32, klo);
      v16h a1 = load_a_frag(sH  + mrow * HSTR, kc * 32, klo);
#pragma unroll
      for (int nt = 0; nt < 4; ++nt) {
        v16h bi = *(const v16h*)(sWi + ((size_t)((wv * 4 + nt) * 4 + kc) * 32 + lane) * 16);
        acc[nt] = wmma_f16(a0, bi, acc[nt]);
        v16h bh = *(const v16h*)(sWh + ((size_t)((wv * 4 + nt) * 4 + kc) * 32 + lane) * 16);
        acc[nt] = wmma_f16(a1, bh, acc[nt]);
      }
    }
#pragma unroll
    for (int nt = 0; nt < 4; ++nt) {
      int N = wv * 64 + nt * 16 + mrow;
      float* gp = sG + N * GSTR + khi * 8;
      v4f lo = {acc[nt][0], acc[nt][1], acc[nt][2], acc[nt][3]};
      v4f hi = {acc[nt][4], acc[nt][5], acc[nt][6], acc[nt][7]};
      *(v4f*)(gp) = lo;
      *(v4f*)(gp + 4) = hi;
    }
    __syncthreads();

#pragma unroll
    for (int r = 0; r < 8; ++r) {
      int e = r * 256 + tid;
      int m = e >> 7, j = e & 127;
      float gi = sG[(j)       * GSTR + m] + sB[j];
      float gf = sG[(128 + j) * GSTR + m] + sB[128 + j];
      float gg = sG[(256 + j) * GSTR + m] + sB[256 + j];
      float go = sG[(384 + j) * GSTR + m] + sB[384 + j];
      float iv = dev_sigm(gi), fv = dev_sigm(gf);
      float gv = dev_tanh(gg), ov = dev_sigm(go);
      float c = fv * creg[r] + iv * gv;
      creg[r] = c;
      float h = ov * dev_tanh(c);
      hreg[r] = h;
      sH[m * HSTR + j] = (_Float16)h;
    }
  }

  // ---- FC head: relu(h_last @ Wfc1^T + bfc1) @ Wfc2^T + bfc2 ----
  __syncthreads();
  float* hbuf = sG;          // [16][128] f32
  float* zbuf = sG + 2048;   // [16][64]  f32
#pragma unroll
  for (int r = 0; r < 8; ++r) hbuf[r * 256 + tid] = hreg[r];
  __syncthreads();
  {
    int u = tid & 63;
    int mb = tid >> 6;
    for (int r = 0; r < 4; ++r) {
      int m = mb * 4 + r;
      float a = bfc1[u];
      for (int k = 0; k < 128; ++k) a += hbuf[m * 128 + k] * Wfc1[u * 128 + k];
      zbuf[m * 64 + u] = fmaxf(a, 0.0f);
    }
  }
  __syncthreads();
  if (tid < 16) {
    float a = bfc2[0];
    for (int u = 0; u < 64; ++u) a += zbuf[tid * 64 + u] * Wfc2[u];
    out[bm + tid] = a;
  }
}

// ---------------------------------------------------------------------------
extern "C" void kernel_launch(void* const* d_in, const int* in_sizes, int n_in,
                              void* d_out, int out_size, void* d_ws, size_t ws_size,
                              hipStream_t stream) {
  (void)in_sizes; (void)n_in; (void)out_size; (void)ws_size;
  const float* x      = (const float*)d_in[0];
  const float* W_emb  = (const float*)d_in[1];
  const float* b_emb  = (const float*)d_in[2];
  const float* W_ih0  = (const float*)d_in[3];
  const float* W_hh0  = (const float*)d_in[4];
  const float* b_ih0  = (const float*)d_in[5];
  const float* b_hh0  = (const float*)d_in[6];
  const float* W_ih1  = (const float*)d_in[7];
  const float* W_hh1  = (const float*)d_in[8];
  const float* b_ih1  = (const float*)d_in[9];
  const float* b_hh1  = (const float*)d_in[10];
  const float* W_fc1  = (const float*)d_in[11];
  const float* b_fc1  = (const float*)d_in[12];
  const float* W_fc2  = (const float*)d_in[13];
  const float* b_fc2  = (const float*)d_in[14];
  char* ws = (char*)d_ws;
  float* out = (float*)d_out;

  const int prep_tasks = WC_HALF + 3 * WM_HALF + 1024;
  lstm_prep<<<(prep_tasks + 255) / 256, 256, 0, stream>>>(
      W_emb, b_emb, W_ih0, W_hh0, b_ih0, b_hh0, W_ih1, W_hh1, b_ih1, b_hh1, ws);

  size_t sh0 = (size_t)(WC_HALF + WM_HALF + 16 * XSTR + 16 * HSTR) * 2 +
               (size_t)(512 * GSTR + 512) * 4;   // ~207 KB
  lstm_pass0<<<kNWG, 256, sh0, stream>>>(x, ws);

  size_t sh1 = (size_t)(2 * WM_HALF + 2 * 16 * HSTR) * 2 +
               (size_t)(512 * GSTR + 512) * 4;   // ~307 KB
  lstm_pass1<<<kNWG, 256, sh1, stream>>>(ws, W_fc1, b_fc1, W_fc2, b_fc2, out);
}
